// Model_62955630625124
// MI455X (gfx1250) — matrix operand order repaired and checked by reading, not compiled, not run on hardware
//
#include <hip/hip_runtime.h>
#include <hip/hip_bf16.h>
#include <math.h>

// ---------------- problem constants ----------------
#define S_LEN   2048
#define HID     2048
#define NH      16
#define Q_LORA  1536
#define KV_LORA 512
#define D_NOPE  128
#define D_ROPE  64
#define D_V     128
#define D_QK    (D_NOPE + D_ROPE)        // 192
#define Q_DIM   (NH * D_QK)              // 3072
#define KV_DIM  (NH * (D_NOPE + D_V))    // 4096
#define O_DIM   (NH * D_V)               // 2048
#define CKV_DIM (KV_LORA + D_ROPE)       // 576

typedef __attribute__((ext_vector_type(16))) __bf16 v16bf;
typedef __attribute__((ext_vector_type(8)))  __bf16 v8bf;
typedef __attribute__((ext_vector_type(8)))  float  v8f;

// ---------------- fragment helpers (ISA 7.12.2 layouts, vectorized) ----------------
__device__ __forceinline__ v16bf concat8(v8bf lo, v8bf hi) {
    v16bf r;
#pragma unroll
    for (int e = 0; e < 8; ++e) { r[e] = lo[e]; r[e + 8] = hi[e]; }
    return r;
}
// A-frag from row-major [16][ld] (ld multiple of 8): two contiguous 8-elem runs
// at (m, half*8) and (m, 16+half*8)  -> two ds_load_b128
__device__ __forceinline__ v16bf lds_a_frag(const __bf16* p, int ld, int lane) {
    int m = lane & 15, half = lane >> 4;
    const __bf16* q = p + m * ld + half * 8;
    return concat8(*(const v8bf*)q, *(const v8bf*)(q + 16));
}
// B-frag from N-major storage [16 n][ld k]: two 8-elem runs at (n, half*8) and (n, 16+half*8)
__device__ __forceinline__ v16bf lds_b_fragT(const __bf16* p, int ld, int lane) {
    int n = lane & 15, half = lane >> 4;
    const __bf16* q = p + n * ld + half * 8;
    return concat8(*(const v8bf*)q, *(const v8bf*)(q + 16));
}
// convert 8 contiguous floats (16B-aligned) -> v8bf
__device__ __forceinline__ v8bf cvt8(const float* s) {
    float4 f0 = *(const float4*)s, f1 = *(const float4*)(s + 4);
    v8bf v;
    v[0] = (__bf16)f0.x; v[1] = (__bf16)f0.y; v[2] = (__bf16)f0.z; v[3] = (__bf16)f0.w;
    v[4] = (__bf16)f1.x; v[5] = (__bf16)f1.y; v[6] = (__bf16)f1.z; v[7] = (__bf16)f1.w;
    return v;
}
__device__ __forceinline__ v8f wmma_bf16(v16bf a, v16bf b, v8f c) {
    return __builtin_amdgcn_wmma_f32_16x16x32_bf16(false, a, false, b, (short)0, c,
                                                   false, false);
}
// async 16B global -> LDS copy (per-lane addresses), tracked by ASYNCcnt
__device__ __forceinline__ void async_g2l_b128(unsigned lds_off, const __bf16* g) {
    unsigned long long ga = (unsigned long long)(size_t)g;
    asm volatile("global_load_async_to_lds_b128 %0, %1, off"
                 :: "v"(lds_off), "v"(ga) : "memory");
}
__device__ __forceinline__ void wait_asynccnt0() {
    asm volatile("s_wait_asynccnt 0x0" ::: "memory");
}

// ---------------- weight convert + transpose: W[K][N] f32 -> Wt[N][K] bf16 ----------------
__global__ __launch_bounds__(256) void transpose_cvt_kernel(
    const float* __restrict__ W, __bf16* __restrict__ Wt, int K, int N) {
    __shared__ float tile[32][33];
    int k0 = blockIdx.x * 32, n0 = blockIdx.y * 32;
    int tx = threadIdx.x & 31, ty = threadIdx.x >> 5;   // 32 x 8
#pragma unroll
    for (int i = ty; i < 32; i += 8)
        tile[i][tx] = W[(size_t)(k0 + i) * N + n0 + tx];
    __syncthreads();
#pragma unroll
    for (int i = ty; i < 32; i += 8)
        Wt[(size_t)(n0 + i) * K + k0 + tx] = (__bf16)tile[tx][i];
}

// ---------------- tiled GEMM: C[M,N] = A[M,K(lda)] * Bt[N][K] ----------------
// 128 threads = 4 waves; block tile 128x64, K-step 32; wave tile 32x64 (8 WMMA/k-step)
// B tiles arrive via global_load_async_to_lds_b128 (bf16, pre-transposed in global)
#define GK_LD 40   // 32 + 8 pad (80B rows, 16B-aligned chunks)
__global__ __launch_bounds__(128) void gemm_bf16_kernel(
    const float* __restrict__ A, const __bf16* __restrict__ Bt, float* __restrict__ C,
    int M, int N, int K, int lda, int ldc) {
    __shared__ __align__(16) __bf16 As[128][GK_LD];   // 10 KB
    __shared__ __align__(16) __bf16 BsT[64][GK_LD];   //  5 KB (N-major)
    int tid = threadIdx.x, lane = tid & 31, wave = tid >> 5;
    int m0 = blockIdx.y * 128, n0 = blockIdx.x * 64;

    v8f acc[2][4];
#pragma unroll
    for (int s = 0; s < 2; ++s)
#pragma unroll
        for (int nt = 0; nt < 4; ++nt) acc[s][nt] = v8f{};

    for (int k0 = 0; k0 < K; k0 += 32) {
        // B tile: 64 rows x 32 k = 256 chunks of 16B, async copied straight to LDS
#pragma unroll
        for (int c = 0; c < 2; ++c) {
            int id = tid + c * 128;                  // 0..255
            int n = id >> 2, kh = (id & 3) * 8;
            async_g2l_b128((unsigned)(size_t)&BsT[n][kh],
                           Bt + (size_t)(n0 + n) * K + k0 + kh);
        }
        // A tile: f32 -> bf16 through VGPRs
        {
            int kh = (tid & 3) * 8;
#pragma unroll
            for (int r = tid >> 2; r < 128; r += 32)
                *(v8bf*)&As[r][kh] = cvt8(A + (size_t)(m0 + r) * lda + k0 + kh);
        }
        wait_asynccnt0();
        __syncthreads();
        v16bf a0 = lds_a_frag(&As[wave * 32][0],      GK_LD, lane);
        v16bf a1 = lds_a_frag(&As[wave * 32 + 16][0], GK_LD, lane);
#pragma unroll
        for (int nt = 0; nt < 4; ++nt) {
            v16bf b = lds_b_fragT(&BsT[nt * 16][0], GK_LD, lane);
            acc[0][nt] = wmma_bf16(a0, b, acc[0][nt]);
            acc[1][nt] = wmma_bf16(a1, b, acc[1][nt]);
        }
        __syncthreads();
    }
    int nl = lane & 15, half = lane >> 4;
#pragma unroll
    for (int s = 0; s < 2; ++s)
#pragma unroll
        for (int nt = 0; nt < 4; ++nt)
#pragma unroll
            for (int r = 0; r < 8; ++r) {
                int m = m0 + wave * 32 + s * 16 + r + half * 8;
                int n = n0 + nt * 16 + nl;
                C[(size_t)m * ldc + n] = acc[s][nt][r];
            }
}

// ---------------- RMSNorm (row-wise), strided in/out ----------------
__global__ __launch_bounds__(256) void rmsnorm_kernel(
    const float* __restrict__ x, const float* __restrict__ w, float* __restrict__ y,
    int D, int xstride, int ystride) {
    int row = blockIdx.x;
    const float* xr = x + (size_t)row * xstride;
    float* yr = y + (size_t)row * ystride;
    float s = 0.f;
    for (int i = threadIdx.x; i < D; i += 256) { float v = xr[i]; s += v * v; }
#pragma unroll
    for (int off = 16; off; off >>= 1) s += __shfl_xor(s, off, 32);
    __shared__ float wsum[8];
    __shared__ float rtot;
    if ((threadIdx.x & 31) == 0) wsum[threadIdx.x >> 5] = s;
    __syncthreads();
    if (threadIdx.x == 0) {
        float t = 0.f;
        for (int i = 0; i < 8; ++i) t += wsum[i];
        rtot = rsqrtf(t / (float)D + 1e-6f);
    }
    __syncthreads();
    float r = rtot;
    for (int i = threadIdx.x; i < D; i += 256) yr[i] = w[i] * (xr[i] * r);
}

// ---------------- RoPE: q_pe (in-place, all heads) and k_pe -> kpe_out ----------------
__global__ __launch_bounds__(64) void rope_kernel(
    float* __restrict__ q, const float* __restrict__ ckv, float* __restrict__ kpe_out) {
    int t = blockIdx.x;
    int i = threadIdx.x;                          // 0..63
    float inv_freq = powf(10000.f, -(float)(2 * (i & 31)) / (float)D_ROPE);
    float ang = (float)t * inv_freq;
    float c = __cosf(ang), sn = __sinf(ang);

    const float* kp = ckv + (size_t)t * CKV_DIM + KV_LORA;
    float krot = (i < 32) ? -kp[i + 32] : kp[i - 32];
    kpe_out[(size_t)t * D_ROPE + i] = kp[i] * c + krot * sn;

    for (int h = 0; h < NH; ++h) {
        float* qp = q + (size_t)t * Q_DIM + h * D_QK + D_NOPE;
        float x = qp[i];
        float rot = (i < 32) ? -qp[i + 32] : qp[i - 32];
        __syncthreads();                          // all reads before in-place writes
        qp[i] = x * c + rot * sn;
        __syncthreads();
    }
}

// ---------------- flash attention, one (64-query tile, head) per block ----------------
#define KS_LD 200   // 192 + 8 pad
#define VS_LD 72    // 64 + 8 pad
#define PS_LD 72
__global__ __launch_bounds__(128) void mla_attn_kernel(
    const float* __restrict__ q, const float* __restrict__ kv,
    const float* __restrict__ kpe, float* __restrict__ out) {
    int qt = blockIdx.x, h = blockIdx.y;
    int tid = threadIdx.x, lane = tid & 31, wave = tid >> 5;
    int nl = lane & 15, half = lane >> 4;
    int q0 = qt * 64;

    __shared__ __align__(16) __bf16 Ks[64][KS_LD];      // [key][d]  (N-major for QK^T)
    __shared__ __align__(16) __bf16 VsT[128][VS_LD];    // [d][key]  (N-major for P*V)
    __shared__ __align__(16) __bf16 Ps[4][16][PS_LD];   // per-wave P staging

    // Q fragments straight from global (float4 reads, A-layout chunks)
    v16bf aq[6];
    {
        const float* qrow = q + (size_t)(q0 + wave * 16 + nl) * Q_DIM + h * D_QK;
#pragma unroll
        for (int ks = 0; ks < 6; ++ks) {
            const float* base = qrow + ks * 32 + half * 8;
            aq[ks] = concat8(cvt8(base), cvt8(base + 16));
        }
    }

    v8f o[8];
#pragma unroll
    for (int dt = 0; dt < 8; ++dt) o[dt] = v8f{};
    float m_i[8], l_i[8];
#pragma unroll
    for (int r = 0; r < 8; ++r) { m_i[r] = -1e30f; l_i[r] = 0.f; }
    const float scale = 0.072168784f;   // 192^-0.5

    for (int kt = 0; kt <= qt; ++kt) {
        int k0 = kt * 64;
        __syncthreads();
        // K tile [key][d]: two threads per key row, 96 d each, 8-float chunks
        {
            int r = tid >> 1, dh = (tid & 1) * 96;
            const float* kvrow = kv + (size_t)(k0 + r) * KV_DIM + h * (D_NOPE + D_V);
            const float* kprow = kpe + (size_t)(k0 + r) * D_ROPE;
#pragma unroll
            for (int c8 = 0; c8 < 96; c8 += 8) {
                int d = dh + c8;
                const float* src = (d < D_NOPE) ? (kvrow + d) : (kprow + d - D_NOPE);
                *(v8bf*)&Ks[r][d] = cvt8(src);
            }
        }
        // V tile transposed [d][key]: one thread per d, coalesced across lanes
        {
            int d = tid;   // 0..127
#pragma unroll
            for (int j0 = 0; j0 < 64; j0 += 8) {
                v8bf v;
#pragma unroll
                for (int j = 0; j < 8; ++j)
                    v[j] = (__bf16)kv[(size_t)(k0 + j0 + j) * KV_DIM +
                                      h * (D_NOPE + D_V) + D_NOPE + d];
                *(v8bf*)&VsT[d][j0] = v;
            }
        }
        __syncthreads();

        // S = Q * K^T  (4 key n-tiles x 6 k-steps)
        v8f sfr[4];
#pragma unroll
        for (int nt = 0; nt < 4; ++nt) sfr[nt] = v8f{};
#pragma unroll
        for (int nt = 0; nt < 4; ++nt)
#pragma unroll
            for (int ks = 0; ks < 6; ++ks) {
                v16bf b = lds_b_fragT(&Ks[nt * 16][ks * 32], KS_LD, lane);
                sfr[nt] = wmma_bf16(aq[ks], b, sfr[nt]);
            }

        // scale + causal mask (diagonal tile only)
#pragma unroll
        for (int nt = 0; nt < 4; ++nt)
#pragma unroll
            for (int r = 0; r < 8; ++r) {
                float s = sfr[nt][r] * scale;
                if (kt == qt) {
                    int qm = q0 + wave * 16 + r + half * 8;
                    int kn = k0 + nt * 16 + nl;
                    if (kn > qm) s = -1e30f;
                }
                sfr[nt][r] = s;
            }

        // online softmax: each row spans the 16 lanes of a half-group
#pragma unroll
        for (int r = 0; r < 8; ++r) {
            float mx = fmaxf(fmaxf(sfr[0][r], sfr[1][r]), fmaxf(sfr[2][r], sfr[3][r]));
#pragma unroll
            for (int off = 8; off; off >>= 1) mx = fmaxf(mx, __shfl_xor(mx, off, 32));
            float mnew = fmaxf(m_i[r], mx);
            float alpha = __expf(m_i[r] - mnew);
            m_i[r] = mnew;
            float rs = 0.f;
#pragma unroll
            for (int nt = 0; nt < 4; ++nt) {
                float p = __expf(sfr[nt][r] - mnew);
                sfr[nt][r] = p;
                rs += p;
            }
#pragma unroll
            for (int off = 8; off; off >>= 1) rs += __shfl_xor(rs, off, 32);
            l_i[r] = l_i[r] * alpha + rs;
#pragma unroll
            for (int dt = 0; dt < 8; ++dt) o[dt][r] *= alpha;
            int mrow = r + half * 8;
#pragma unroll
            for (int nt = 0; nt < 4; ++nt)
                Ps[wave][mrow][nt * 16 + nl] = (__bf16)sfr[nt][r];
        }

        // O += P * V   (K=64 -> 2 k-steps; 128 d -> 8 n-tiles)
        v16bf ap0 = lds_a_frag(&Ps[wave][0][0],  PS_LD, lane);
        v16bf ap1 = lds_a_frag(&Ps[wave][0][32], PS_LD, lane);
#pragma unroll
        for (int dt = 0; dt < 8; ++dt) {
            v16bf b0 = lds_b_fragT(&VsT[dt * 16][0],  VS_LD, lane);
            v16bf b1 = lds_b_fragT(&VsT[dt * 16][32], VS_LD, lane);
            o[dt] = wmma_bf16(ap0, b0, o[dt]);
            o[dt] = wmma_bf16(ap1, b1, o[dt]);
        }
    }

    // epilogue: O / l -> out[q, h*128 + d]
#pragma unroll
    for (int r = 0; r < 8; ++r) {
        float inv = 1.0f / l_i[r];
        int qm = q0 + wave * 16 + r + half * 8;
#pragma unroll
        for (int dt = 0; dt < 8; ++dt)
            out[(size_t)qm * O_DIM + h * D_V + dt * 16 + nl] = o[dt][r] * inv;
    }
}

// ---------------- host-side orchestration ----------------
extern "C" void kernel_launch(void* const* d_in, const int* in_sizes, int n_in,
                              void* d_out, int out_size, void* d_ws, size_t ws_size,
                              hipStream_t stream) {
    const float* hidden  = (const float*)d_in[0];   // [2048, 2048]
    const float* Wqa     = (const float*)d_in[1];   // [2048, 1536]
    const float* q_ln_w  = (const float*)d_in[2];   // [1536]
    const float* Wqb     = (const float*)d_in[3];   // [1536, 3072]
    const float* Wkva    = (const float*)d_in[4];   // [2048, 576]
    const float* kv_ln_w = (const float*)d_in[5];   // [512]
    const float* Wkvb    = (const float*)d_in[6];   // [512, 4096]
    const float* Wo      = (const float*)d_in[7];   // [2048, 2048]
    float* out = (float*)d_out;

    float* ws = (float*)d_ws;
    float* qa  = ws;                         ws += (size_t)S_LEN * Q_LORA;   // 2048x1536
    float* qm  = ws;                         ws += (size_t)S_LEN * Q_DIM;    // 2048x3072
    float* ckv = ws;                         ws += (size_t)S_LEN * CKV_DIM;  // 2048x576
    float* kvn = ws;                         ws += (size_t)S_LEN * KV_LORA;  // 2048x512
    float* kvm = ws;                         ws += (size_t)S_LEN * KV_DIM;   // 2048x4096
    float* kpe = ws;                         ws += (size_t)S_LEN * D_ROPE;   // 2048x64
    float* att = ws;                         ws += (size_t)S_LEN * O_DIM;    // 2048x2048
    // bf16 transposed weights (N-major) for async-to-LDS B tiles
    __bf16* wb = (__bf16*)ws;
    __bf16* WqaT  = wb;  wb += (size_t)Q_LORA * HID;     // [1536][2048]
    __bf16* WqbT  = wb;  wb += (size_t)Q_DIM  * Q_LORA;  // [3072][1536]
    __bf16* WkvaT = wb;  wb += (size_t)CKV_DIM * HID;    // [576][2048]
    __bf16* WkvbT = wb;  wb += (size_t)KV_DIM * KV_LORA; // [4096][512]
    __bf16* WoT   = wb;  wb += (size_t)HID * O_DIM;      // [2048][2048]

    dim3 blk(128);
    // 0) weight convert + transpose (f32 [K][N] -> bf16 [N][K])
    transpose_cvt_kernel<<<dim3(HID / 32,    Q_LORA / 32), 256, 0, stream>>>(Wqa,  WqaT,  HID,    Q_LORA);
    transpose_cvt_kernel<<<dim3(Q_LORA / 32, Q_DIM / 32),  256, 0, stream>>>(Wqb,  WqbT,  Q_LORA, Q_DIM);
    transpose_cvt_kernel<<<dim3(HID / 32,    CKV_DIM / 32),256, 0, stream>>>(Wkva, WkvaT, HID,    CKV_DIM);
    transpose_cvt_kernel<<<dim3(KV_LORA / 32,KV_DIM / 32), 256, 0, stream>>>(Wkvb, WkvbT, KV_LORA,KV_DIM);
    transpose_cvt_kernel<<<dim3(O_DIM / 32,  HID / 32),    256, 0, stream>>>(Wo,   WoT,   O_DIM,  HID);

    // 1) qa = hidden @ Wqa
    gemm_bf16_kernel<<<dim3(Q_LORA / 64, S_LEN / 128), blk, 0, stream>>>(
        hidden, WqaT, qa, S_LEN, Q_LORA, HID, HID, Q_LORA);
    // 2) qa = rmsnorm(qa) (in-place)
    rmsnorm_kernel<<<S_LEN, 256, 0, stream>>>(qa, q_ln_w, qa, Q_LORA, Q_LORA, Q_LORA);
    // 3) q = qa @ Wqb
    gemm_bf16_kernel<<<dim3(Q_DIM / 64, S_LEN / 128), blk, 0, stream>>>(
        qa, WqbT, qm, S_LEN, Q_DIM, Q_LORA, Q_LORA, Q_DIM);
    // 4) ckv = hidden @ Wkva
    gemm_bf16_kernel<<<dim3(CKV_DIM / 64, S_LEN / 128), blk, 0, stream>>>(
        hidden, WkvaT, ckv, S_LEN, CKV_DIM, HID, HID, CKV_DIM);
    // 5) kvn = rmsnorm(ckv[:, :512])
    rmsnorm_kernel<<<S_LEN, 256, 0, stream>>>(ckv, kv_ln_w, kvn, KV_LORA, CKV_DIM, KV_LORA);
    // 6) kv = kvn @ Wkvb
    gemm_bf16_kernel<<<dim3(KV_DIM / 64, S_LEN / 128), blk, 0, stream>>>(
        kvn, WkvbT, kvm, S_LEN, KV_DIM, KV_LORA, KV_LORA, KV_DIM);
    // 7) RoPE on q_pe (in-place) and k_pe -> kpe
    rope_kernel<<<S_LEN, 64, 0, stream>>>(qm, ckv, kpe);
    // 8) flash attention per (query tile, head)
    mla_attn_kernel<<<dim3(S_LEN / 64, NH), blk, 0, stream>>>(qm, kvm, kpe, att);
    // 9) out = att @ Wo
    gemm_bf16_kernel<<<dim3(HID / 64, S_LEN / 128), blk, 0, stream>>>(
        att, WoT, out, S_LEN, HID, O_DIM, O_DIM, HID);
}
